// Q6Model_59030030516993
// MI455X (gfx1250) — compile-verified, matrix operand order checked by reading, not run on hardware
//
#include <hip/hip_runtime.h>
#include <hip/hip_bf16.h>
#include <stdint.h>

// ---------------------------------------------------------------------------
// CDNA5 (gfx1250) wave32 WMMA bf16 implementation of the ConvLSTM stack.
// Round 3: hot-loop staging rebuilt -- hoisted/incremental index math,
// b128 vectorized global->LDS staging, K-pair-packed B tile (conflict-free
// dword fragment reads). Inner loop is now loads + v_wmma only.
// ---------------------------------------------------------------------------

typedef __attribute__((ext_vector_type(16))) __bf16        v16bf;
typedef __attribute__((ext_vector_type(8)))  float         v8f;
typedef __attribute__((ext_vector_type(8)))  unsigned int  v8u;

static_assert(sizeof(v16bf) == 32, "v16bf must be 32B");
static_assert(sizeof(v8u) == 32, "v8u must be 32B");

__device__ __forceinline__ uint16_t f2bf(float f) {
  uint32_t u = __float_as_uint(f);
  uint32_t r = u + 0x7FFFu + ((u >> 16) & 1u);   // round-to-nearest-even
  return (uint16_t)(r >> 16);
}
__device__ __forceinline__ float bf2f(uint16_t u) {
  return __uint_as_float(((uint32_t)u) << 16);
}

__global__ void cvt_f32_to_bf16(const float* __restrict__ in,
                                uint16_t* __restrict__ out, int n) {
  int i = blockIdx.x * blockDim.x + threadIdx.x;
  if (i < n) out[i] = f2bf(in[i]);
}

__global__ void zero_u32(uint32_t* __restrict__ p, int n) {
  int i = blockIdx.x * blockDim.x + threadIdx.x;
  if (i < n) p[i] = 0u;
}

// WMMA 16-bit fragment K offsets (per ISA A/B VGPR layout), and the packed
// K-pair index used by the dword-packed B tile.
__device__ __forceinline__ int frag_k0(int v, int half) {
  return ((v < 4) ? (2 * v) : (16 + 2 * (v - 4))) + 8 * half;
}
__device__ __forceinline__ int frag_k2(int v, int half) {
  return ((v < 4) ? v : (8 + (v - 4))) + 4 * half;
}

// interleave low halves of two dwords (two bf16 rows -> one k-pair dword)
__device__ __forceinline__ void pack2(uint32_t a, uint32_t b,
                                      uint32_t& lo, uint32_t& hi) {
  lo = (a & 0xFFFFu) | (b << 16);
  hi = (a >> 16) | (b & 0xFFFF0000u);
}

// ---------------------------------------------------------------------------
// ConvLSTM gate pre-activation: gates = conv(x, Wx, VALID) + conv(h, Wh, SAME) + b
// Implicit GEMM:  M = B*Ho*Wo pixels,  N = 4F gate channels,  K = 9*Cin (+ 9*F).
// Block: 256 threads = 8 waves (4M x 2N). Tile: M=64 x N=128, K stepped by 32.
// ---------------------------------------------------------------------------
__global__ void __launch_bounds__(256)
convlstm_gates_wmma(const uint16_t* __restrict__ xb, int xBatchStride,
                    const uint16_t* __restrict__ hb, int hBatchStride,
                    const uint16_t* __restrict__ Wxw,
                    const uint16_t* __restrict__ Whw,
                    const float* __restrict__ bias,
                    float* __restrict__ gates,
                    int Bsz, int Hin, int Win, int Cin,
                    int Ho, int Wo, int F) {
  const int N4   = 4 * F;
  const int Mtot = Bsz * Ho * Wo;
  const int tid  = threadIdx.x;
  const int wave = tid >> 5;
  const int lane = tid & 31;
  const int half = lane >> 4;
  const int lm   = lane & 15;
  const int msub = (wave >> 1) * 16;
  const int nsub = (wave & 1) * 64;
  const int mbase  = blockIdx.x * 64;
  const int nblock = blockIdx.y * 128;

  // As rows padded to 40 halves (80B, 16B aligned); Bs2 rows padded to 132
  // dwords (16B aligned, cross-half fragment reads land on disjoint banks).
  __shared__ __align__(16) uint16_t As[64][40];
  __shared__ __align__(16) uint32_t Bs2[16][132];

  v8f acc[4] = {};

  // ---- hoisted per-thread staging assignments ----
  // A fast path: thread stages one 16B chunk of row am.
  const int am    = tid >> 2;          // 0..63
  const int apart = (tid & 3) * 8;     // halfword offset in row
  const int apix  = mbase + am;
  const int avalid = apix < Mtot;
  int ab = 0, aoy = 0, aox = 0;
  if (avalid) {
    ab = apix / (Ho * Wo);
    int r = apix - ab * (Ho * Wo);
    aoy = r / Wo;
    aox = r - aoy * Wo;
  }
  // B path: thread stages rows (2*bk2, 2*bk2+1) for 8 columns from bn0.
  const int bk2 = tid >> 4;            // 0..15
  const int bn0 = (tid & 15) * 8;      // 0..120

  for (int phase = 0; phase < 2; ++phase) {
    const int C    = phase ? F : Cin;
    const int Ktot = 9 * C;
    const uint16_t* Wp = phase ? Whw : Wxw;
    const uint16_t* Sp = phase ? hb : xb;
    const int SBS = phase ? hBatchStride : xBatchStride;
    const int SW  = phase ? Wo : Win;
    const bool fastA = (C & 31) == 0;  // 32-wide K step stays inside one tap

    int ky = 0, kx = 0, ci0 = 0;       // incremental k decomposition (fastA)

    for (int kb = 0; kb < Ktot; kb += 32) {
      if (fastA) {
        // ---- A: one contiguous 16B chunk per thread ----
        uint4 av = {0, 0, 0, 0};
        int iy, ix, ok = avalid;
        if (phase == 0) { iy = aoy + ky;     ix = aox + kx; }
        else           { iy = aoy + ky - 1; ix = aox + kx - 1;
                         ok = ok && iy >= 0 && iy < Ho && ix >= 0 && ix < Wo; }
        if (ok)
          av = *(const uint4*)&Sp[(size_t)ab * SBS +
                                  ((size_t)iy * SW + ix) * C + ci0 + apart];
        *(uint4*)&As[am][apart] = av;

        // ---- B: two weight rows, packed into k-pair dwords ----
        const uint16_t* r0 = Wp + (size_t)(kb + 2 * bk2) * N4 + nblock + bn0;
        uint4 w0 = *(const uint4*)r0;
        uint4 w1 = *(const uint4*)(r0 + N4);
        uint4 s0, s1;
        pack2(w0.x, w1.x, s0.x, s0.y);
        pack2(w0.y, w1.y, s0.z, s0.w);
        pack2(w0.z, w1.z, s1.x, s1.y);
        pack2(w0.w, w1.w, s1.z, s1.w);
        *(uint4*)&Bs2[bk2][bn0]     = s0;
        *(uint4*)&Bs2[bk2][bn0 + 4] = s1;
      } else {
        // ---- generic scalar staging (layer-1 input conv only, K=9) ----
        for (int e = tid; e < 64 * 32; e += 256) {
          int m = e >> 5, k = e & 31;
          uint16_t val = 0;
          int pixel = mbase + m, kk = kb + k;
          if (pixel < Mtot && kk < Ktot) {
            int b  = pixel / (Ho * Wo);
            int r  = pixel - b * (Ho * Wo);
            int oy = r / Wo, ox = r - (r / Wo) * Wo;
            int kyg = kk / (3 * C);
            int r2  = kk - kyg * 3 * C;
            int kxg = r2 / C;
            int ci  = r2 - kxg * C;
            if (phase == 0) {
              val = Sp[(size_t)b * SBS + ((size_t)(oy + kyg) * SW + ox + kxg) * C + ci];
            } else {
              int iy = oy + kyg - 1, ix = ox + kxg - 1;
              if (iy >= 0 && iy < Ho && ix >= 0 && ix < Wo)
                val = Sp[(size_t)b * SBS + ((size_t)iy * SW + ix) * C + ci];
            }
          }
          As[m][k] = val;
        }
        for (int e = tid; e < 16 * 128; e += 256) {
          int k2 = e >> 7, n = e & 127;
          int k0r = kb + 2 * k2;
          uint32_t a = (k0r < Ktot) ? Wp[(size_t)k0r * N4 + nblock + n] : 0u;
          uint32_t b = (k0r + 1 < Ktot) ? Wp[(size_t)(k0r + 1) * N4 + nblock + n] : 0u;
          Bs2[k2][n] = a | (b << 16);
        }
      }
      __syncthreads();

      // ---- fragments + WMMA ----
      v8u au;
#pragma unroll
      for (int v = 0; v < 8; ++v)
        au[v] = *(const uint32_t*)&As[msub + lm][frag_k0(v, half)];
      v16bf af = __builtin_bit_cast(v16bf, au);
#pragma unroll
      for (int j = 0; j < 4; ++j) {
        int n = nsub + j * 16 + lm;
        v8u bu;
#pragma unroll
        for (int v = 0; v < 8; ++v)
          bu[v] = Bs2[frag_k2(v, half)][n];
        v16bf bfv = __builtin_bit_cast(v16bf, bu);
        acc[j] = __builtin_amdgcn_wmma_f32_16x16x32_bf16(
            false, af, false, bfv, (short)0, acc[j], false, false);
      }
      __syncthreads();

      if (fastA) {  // 3-op carry instead of divides
        ci0 += 32;
        if (ci0 >= C) { ci0 = 0; ++kx; if (kx == 3) { kx = 0; ++ky; } }
      }
    }
  }

  // ---- epilogue: + bias, store f32 gates ----
#pragma unroll
  for (int v = 0; v < 8; ++v) {
    int pixel = mbase + msub + v + 8 * half;
    if (pixel < Mtot) {
#pragma unroll
      for (int j = 0; j < 4; ++j) {
        int ng = nblock + nsub + j * 16 + lm;
        gates[(size_t)pixel * N4 + ng] = acc[j][v] + bias[ng];
      }
    }
  }
}

// ---------------------------------------------------------------------------
// LSTM elementwise: i,f,g,o gates -> c,h ; h re-emitted as bf16 for next GEMMs
// ---------------------------------------------------------------------------
__global__ void lstm_pointwise(const float* __restrict__ gates,
                               float* __restrict__ c,
                               uint16_t* __restrict__ h_bf,
                               int Npix, int F) {
  int idx = blockIdx.x * blockDim.x + threadIdx.x;
  if (idx >= Npix * F) return;
  int p = idx / F, f = idx - p * F;
  const float* g = gates + (size_t)p * 4 * F;
  float ig = 1.f / (1.f + __expf(-g[f]));
  float fg = 1.f / (1.f + __expf(-g[F + f]));
  float gg = tanhf(g[2 * F + f]);
  float og = 1.f / (1.f + __expf(-g[3 * F + f]));
  float cn = fg * c[idx] + ig * gg;
  c[idx] = cn;
  h_bf[idx] = f2bf(og * tanhf(cn));
}

// ---------------------------------------------------------------------------
// Plain WMMA GEMM: out = act(A @ W + b), A (MxK) bf16, W (KxN) bf16, out bf16.
// Requires N % 8 == 0 (true for all uses: 256, 64). Same 64x128 block tile.
// ---------------------------------------------------------------------------
__global__ void __launch_bounds__(256)
gemm_bias_relu_wmma(const uint16_t* __restrict__ A,
                    const uint16_t* __restrict__ W,
                    const float* __restrict__ bias,
                    uint16_t* __restrict__ outb,
                    int M, int K, int N, int doRelu) {
  const int tid  = threadIdx.x;
  const int wave = tid >> 5;
  const int lane = tid & 31;
  const int half = lane >> 4;
  const int lm   = lane & 15;
  const int msub = (wave >> 1) * 16;
  const int nsub = (wave & 1) * 64;
  const int mbase  = blockIdx.x * 64;
  const int nblock = blockIdx.y * 128;

  __shared__ __align__(16) uint16_t As[64][40];
  __shared__ __align__(16) uint32_t Bs2[16][132];

  const int am    = tid >> 2;
  const int apart = (tid & 3) * 8;
  const int apix  = mbase + am;
  const int bk2 = tid >> 4;
  const int bn0 = (tid & 15) * 8;
  const int bngood = (nblock + bn0) < N;   // whole 8-col chunk in range

  v8f acc[4] = {};

  for (int kb = 0; kb < K; kb += 32) {
    // ---- A: contiguous 16B chunk per thread ----
    uint4 av = {0, 0, 0, 0};
    if (apix < M) {
      int kbs = kb + apart;
      if (kbs + 8 <= K) {
        av = *(const uint4*)&A[(size_t)apix * K + kbs];
      } else {                                  // ragged K tail (unused here)
        uint32_t d0 = 0, d1 = 0, d2 = 0, d3 = 0;
#pragma unroll
        for (int i = 0; i < 8; ++i)
          if (kbs + i < K) {
            uint32_t xv = A[(size_t)apix * K + kbs + i];
            uint32_t sh = (uint32_t)(i & 1) * 16;
            if (i < 2) d0 |= xv << sh;
            else if (i < 4) d1 |= xv << sh;
            else if (i < 6) d2 |= xv << sh;
            else d3 |= xv << sh;
          }
        av.x = d0; av.y = d1; av.z = d2; av.w = d3;
      }
    }
    *(uint4*)&As[am][apart] = av;

    // ---- B: two weight rows packed into k-pair dwords ----
    uint4 w0 = {0, 0, 0, 0}, w1 = {0, 0, 0, 0};
    int k0r = kb + 2 * bk2;
    if (bngood) {
      if (k0r < K)     w0 = *(const uint4*)&W[(size_t)k0r * N + nblock + bn0];
      if (k0r + 1 < K) w1 = *(const uint4*)&W[(size_t)(k0r + 1) * N + nblock + bn0];
    }
    uint4 s0, s1;
    pack2(w0.x, w1.x, s0.x, s0.y);
    pack2(w0.y, w1.y, s0.z, s0.w);
    pack2(w0.z, w1.z, s1.x, s1.y);
    pack2(w0.w, w1.w, s1.z, s1.w);
    *(uint4*)&Bs2[bk2][bn0]     = s0;
    *(uint4*)&Bs2[bk2][bn0 + 4] = s1;
    __syncthreads();

    v8u au;
#pragma unroll
    for (int v = 0; v < 8; ++v)
      au[v] = *(const uint32_t*)&As[msub + lm][frag_k0(v, half)];
    v16bf af = __builtin_bit_cast(v16bf, au);
#pragma unroll
    for (int j = 0; j < 4; ++j) {
      int n = nsub + j * 16 + lm;
      v8u bu;
#pragma unroll
      for (int v = 0; v < 8; ++v)
        bu[v] = Bs2[frag_k2(v, half)][n];
      v16bf bfv = __builtin_bit_cast(v16bf, bu);
      acc[j] = __builtin_amdgcn_wmma_f32_16x16x32_bf16(
          false, af, false, bfv, (short)0, acc[j], false, false);
    }
    __syncthreads();
  }

#pragma unroll
  for (int v = 0; v < 8; ++v) {
    int pixel = mbase + msub + v + 8 * half;
    if (pixel < M) {
#pragma unroll
      for (int j = 0; j < 4; ++j) {
        int ng = nblock + nsub + j * 16 + lm;
        if (ng < N) {
          float val = acc[j][v] + bias[ng];
          if (doRelu) val = fmaxf(val, 0.f);
          outb[(size_t)pixel * N + ng] = f2bf(val);
        }
      }
    }
  }
}

// ---------------------------------------------------------------------------
// concat([h(B,28,28,64), t(B,1,28,64), loc(B,1,28,64)], axis=1) -> bf16
// ---------------------------------------------------------------------------
__global__ void concat_rows(const uint16_t* __restrict__ h,
                            const float* __restrict__ tin,
                            const float* __restrict__ loc,
                            uint16_t* __restrict__ outb, int Bsz) {
  int idx = blockIdx.x * blockDim.x + threadIdx.x;
  int total = Bsz * 30 * 28 * 64;
  if (idx >= total) return;
  int ch = idx & 63;
  int t2 = idx >> 6;
  int cp = t2 % 28; t2 /= 28;
  int r  = t2 % 30;
  int b  = t2 / 30;
  uint16_t v;
  if (r < 28)        v = h[((size_t)(b * 28 + r) * 28 + cp) * 64 + ch];
  else if (r == 28)  v = f2bf(tin[((size_t)b * 28 + cp) * 64 + ch]);
  else               v = f2bf(loc[((size_t)b * 28 + cp) * 64 + ch]);
  outb[idx] = v;
}

// final 64 -> 1 projection (too thin for WMMA): out = fc1 @ Wf2 + bf2
__global__ void final_head(const uint16_t* __restrict__ a,
                           const float* __restrict__ w,
                           const float* __restrict__ b,
                           float* __restrict__ out, int M) {
  int i = blockIdx.x * blockDim.x + threadIdx.x;
  if (i >= M) return;
  float s = 0.f;
#pragma unroll
  for (int k = 0; k < 64; ++k) s += bf2f(a[(size_t)i * 64 + k]) * w[k];
  out[i] = s + b[0];
}

// ---------------------------------------------------------------------------
extern "C" void kernel_launch(void* const* d_in, const int* in_sizes, int n_in,
                              void* d_out, int out_size, void* d_ws, size_t ws_size,
                              hipStream_t stream) {
  (void)in_sizes; (void)n_in; (void)out_size; (void)ws_size;

  const float* x    = (const float*)d_in[0];
  const float* t_in = (const float*)d_in[1];
  const float* loc  = (const float*)d_in[2];
  const float* Wx1  = (const float*)d_in[3];
  const float* Wh1  = (const float*)d_in[4];
  const float* b1   = (const float*)d_in[5];
  const float* Wx2  = (const float*)d_in[6];
  const float* Wh2  = (const float*)d_in[7];
  const float* b2   = (const float*)d_in[8];
  const float* Wd1  = (const float*)d_in[9];
  const float* bd1  = (const float*)d_in[10];
  const float* Wd2  = (const float*)d_in[11];
  const float* bd2  = (const float*)d_in[12];
  const float* Wf1  = (const float*)d_in[13];
  const float* bf1  = (const float*)d_in[14];
  const float* Wf2  = (const float*)d_in[15];
  const float* bf2  = (const float*)d_in[16];
  float* out = (float*)d_out;

  const int Bsz = 8, T = 10, E = 32, F1 = 256, F2 = 128;
  const int H1 = 30, M1 = Bsz * H1 * H1;   // 7200 pixels, layer-1 output
  const int H2 = 28, M2 = Bsz * H2 * H2;   // 6272 pixels, layer-2 output

  // ---- carve workspace ----
  char* base = (char*)d_ws;
  size_t off = 0;
  auto carve = [&](size_t bytes) -> void* {
    void* p = base + off;
    off = (off + bytes + 255) & ~(size_t)255;
    return p;
  };
  uint16_t* x_bf = (uint16_t*)carve((size_t)Bsz * T * E * E * 2);
  uint16_t* Wx1b = (uint16_t*)carve((size_t)9 * 4 * F1 * 2);
  uint16_t* Wh1b = (uint16_t*)carve((size_t)9 * F1 * 4 * F1 * 2);
  uint16_t* Wx2b = (uint16_t*)carve((size_t)9 * F1 * 4 * F2 * 2);
  uint16_t* Wh2b = (uint16_t*)carve((size_t)9 * F2 * 4 * F2 * 2);
  uint16_t* Wd1b = (uint16_t*)carve((size_t)F2 * 256 * 2);
  uint16_t* Wd2b = (uint16_t*)carve((size_t)256 * 64 * 2);
  uint16_t* Wf1b = (uint16_t*)carve((size_t)64 * 64 * 2);
  uint16_t* h1b  = (uint16_t*)carve((size_t)M1 * F1 * 2);
  float*    c1   = (float*)   carve((size_t)M1 * F1 * 4);
  float*    g1   = (float*)   carve((size_t)M1 * 4 * F1 * 4);
  uint16_t* h2b  = (uint16_t*)carve((size_t)M2 * F2 * 2);
  float*    c2   = (float*)   carve((size_t)M2 * F2 * 4);
  float*    g2   = (float*)   carve((size_t)M2 * 4 * F2 * 4);
  uint16_t* d1b  = (uint16_t*)carve((size_t)M2 * 256 * 2);
  uint16_t* d2b  = (uint16_t*)carve((size_t)M2 * 64 * 2);
  uint16_t* catb = (uint16_t*)carve((size_t)Bsz * 30 * 28 * 64 * 2);
  uint16_t* fc1b = (uint16_t*)carve((size_t)Bsz * 30 * 28 * 64 * 2);

  auto cvt = [&](const float* src, uint16_t* dst, int n) {
    cvt_f32_to_bf16<<<(n + 255) / 256, 256, 0, stream>>>(src, dst, n);
  };
  cvt(x,   x_bf, Bsz * T * E * E);
  cvt(Wx1, Wx1b, 9 * 4 * F1);
  cvt(Wh1, Wh1b, 9 * F1 * 4 * F1);
  cvt(Wx2, Wx2b, 9 * F1 * 4 * F2);
  cvt(Wh2, Wh2b, 9 * F2 * 4 * F2);
  cvt(Wd1, Wd1b, F2 * 256);
  cvt(Wd2, Wd2b, 256 * 64);
  cvt(Wf1, Wf1b, 64 * 64);

  auto zero = [&](void* p, size_t bytes) {
    int n = (int)(bytes / 4);
    zero_u32<<<(n + 255) / 256, 256, 0, stream>>>((uint32_t*)p, n);
  };
  zero(h1b, (size_t)M1 * F1 * 2);
  zero(c1,  (size_t)M1 * F1 * 4);
  zero(h2b, (size_t)M2 * F2 * 2);
  zero(c2,  (size_t)M2 * F2 * 4);

  // ---- interleaved ConvLSTM timesteps ----
  for (int t = 0; t < T; ++t) {
    convlstm_gates_wmma<<<dim3((M1 + 63) / 64, (4 * F1) / 128), 256, 0, stream>>>(
        x_bf + (size_t)t * E * E, T * E * E,
        h1b, H1 * H1 * F1, Wx1b, Wh1b, b1, g1,
        Bsz, E, E, 1, H1, H1, F1);
    lstm_pointwise<<<(M1 * F1 + 255) / 256, 256, 0, stream>>>(g1, c1, h1b, M1, F1);

    convlstm_gates_wmma<<<dim3((M2 + 63) / 64, (4 * F2) / 128), 256, 0, stream>>>(
        h1b, H1 * H1 * F1,
        h2b, H2 * H2 * F2, Wx2b, Wh2b, b2, g2,
        Bsz, H1, H1, F1, H2, H2, F2);
    lstm_pointwise<<<(M2 * F2 + 255) / 256, 256, 0, stream>>>(g2, c2, h2b, M2, F2);
  }

  // ---- dense head ----
  gemm_bias_relu_wmma<<<dim3((M2 + 63) / 64, 2), 256, 0, stream>>>(
      h2b, Wd1b, bd1, d1b, M2, F2, 256, 1);
  gemm_bias_relu_wmma<<<dim3((M2 + 63) / 64, 1), 256, 0, stream>>>(
      d1b, Wd2b, bd2, d2b, M2, 256, 64, 1);

  int ncat = Bsz * 30 * 28 * 64;
  concat_rows<<<(ncat + 255) / 256, 256, 0, stream>>>(d2b, t_in, loc, catb, Bsz);

  int Mc = Bsz * 30 * 28;
  gemm_bias_relu_wmma<<<dim3((Mc + 63) / 64, 1), 256, 0, stream>>>(
      catb, Wf1b, bf1, fc1b, Mc, 64, 64, 1);
  final_head<<<(Mc + 255) / 256, 256, 0, stream>>>(fc1b, Wf2, bf2, out, Mc);
}
